// _MambaLikeBlock_84782654423224
// MI455X (gfx1250) — matrix operand order repaired and checked by reading, not compiled, not run on hardware
//
#include <hip/hip_runtime.h>

// ---------------------------------------------------------------------------
// MambaLikeBlock for MI455X (gfx1250, wave32, WMMA).
// GEMMs: f16 WMMA w/ f32 accumulation; weights pre-transposed to f16 (fit in
// 192MB L2). LDS staging via GLOBAL_LOAD_ASYNC_TO_LDS_B128, double-buffered
// 64-K slabs (ASYNCcnt-pipelined, branch-free steady state).
// Scan: 3-pass chunked linear recurrence.
// ---------------------------------------------------------------------------

typedef _Float16 h16;
typedef __attribute__((ext_vector_type(16))) _Float16 v16h;
typedef __attribute__((ext_vector_type(8)))  _Float16 v8h;
typedef __attribute__((ext_vector_type(8)))  float    v8f;

union V16U { v16h v; v8h h[2]; };

#define TSEQ 8192
#define DMOD 2048
#define NCHUNK 64
#define CHLEN 128   // TSEQ / NCHUNK

__device__ __forceinline__ float sigmoidf_(float v) {
    return 1.0f / (1.0f + __expf(-v));
}

// Async global->LDS copy, 16 bytes per lane (ASYNCcnt-tracked).
__device__ __forceinline__ void async_copy_b128(unsigned lds_off,
                                                unsigned long long gaddr) {
    asm volatile("global_load_async_to_lds_b128 %0, %1, off"
                 :: "v"(lds_off), "v"(gaddr) : "memory");
}

// ---------------------------------------------------------------------------
// LayerNorm over rows of x[T,D], emit f16 (WMMA A-operand).
// ---------------------------------------------------------------------------
__global__ __launch_bounds__(256)
void ln_to_f16_kernel(const float* __restrict__ x,
                      const float* __restrict__ g,
                      const float* __restrict__ b,
                      h16* __restrict__ out, int D)
{
    __shared__ float rs[256];
    __shared__ float rs2[256];
    const int row = blockIdx.x;
    const int tid = threadIdx.x;
    const float* xr = x + (size_t)row * D;

    float s = 0.f, s2 = 0.f;
    for (int i = tid; i < D; i += 256) {
        float v = xr[i];
        s += v; s2 += v * v;
    }
    rs[tid] = s; rs2[tid] = s2;
    __syncthreads();
    for (int off = 128; off > 0; off >>= 1) {
        if (tid < off) { rs[tid] += rs[tid + off]; rs2[tid] += rs2[tid + off]; }
        __syncthreads();
    }
    const float mean = rs[0] / (float)D;
    const float var  = rs2[0] / (float)D - mean * mean;
    const float rstd = rsqrtf(var + 1e-5f);

    h16* orow = out + (size_t)row * D;
    for (int i = tid; i < D; i += 256) {
        float v = (xr[i] - mean) * rstd * g[i] + b[i];
        orow[i] = (h16)v;
    }
}

// ---------------------------------------------------------------------------
// W[K][N] f32 -> WT[N][K] f16 (LDS-tiled transpose + convert).
// ---------------------------------------------------------------------------
__global__ __launch_bounds__(256)
void transpose_to_f16_kernel(const float* __restrict__ W, h16* __restrict__ WT,
                             int K, int N)
{
    __shared__ h16 tile[32][33];
    const int tx = threadIdx.x & 31;
    const int ty = threadIdx.x >> 5;
    const int n0 = blockIdx.x * 32;
    const int k0 = blockIdx.y * 32;
    #pragma unroll
    for (int i = 0; i < 32; i += 8)
        tile[ty + i][tx] = (h16)W[(size_t)(k0 + ty + i) * N + n0 + tx];
    __syncthreads();
    #pragma unroll
    for (int i = 0; i < 32; i += 8)
        WT[(size_t)(n0 + ty + i) * K + k0 + tx] = tile[tx][ty + i];
}

// ---------------------------------------------------------------------------
// Tiled WMMA GEMM: C[M,N] = act(A[M,K] @ BT[N,K]^T + bias[N]) (+ res[M,N])
//   A: f16 [M,K] row-major. BT: f16 [N,K] row-major (pre-transposed B).
// Block tile 128x128, staged slab BK=64, 8 waves, wave tile 64x32.
// Per staged slab: 2 sub-steps x 8 WMMA (v_wmma_f32_16x16x32_f16).
// Double-buffered LDS (2 x 32KB), async b128 staging, XOR buffer toggle,
// peeled final iteration (branch-free steady state, constant asynccnt wait).
// M%128==0, N%128==0, K%128==0 assumed.
// ---------------------------------------------------------------------------
template<int ACT, bool OUT_F16, bool HAS_RES>
__global__ __launch_bounds__(256)
void gemm_wmma_kernel(const h16* __restrict__ A, const h16* __restrict__ BT,
                      const float* __restrict__ bias,
                      const float* __restrict__ res,
                      void* __restrict__ Cout, int M, int N, int K)
{
    // buffer: A 128x64 halves (16KB) + B 128x64 (16KB); two buffers = 64KB
    __shared__ h16 smem[2 * 16384];

    const int tid  = threadIdx.x;
    const int lane = tid & 31;
    const int wave = tid >> 5;
    const int bm = blockIdx.y * 128;
    const int bn = blockIdx.x * 128;
    const int wm = (wave >> 2) * 64;
    const int wn = (wave & 3) * 32;
    const int lhalf = lane & 15;
    const int lhi   = lane >> 4;

    // staging: per operand 128 rows x 128B = 1024 b128 chunks; 4 per thread.
    // idx = tid + i*256 (i=0..3): r = idx>>3 (row), q = idx&7 (16B chunk)
    const int r0 = tid >> 3, q0 = tid & 7;

    unsigned long long gaA[4], gaB[4];
    unsigned ldsA[4], ldsB[4];
    #pragma unroll
    for (int i = 0; i < 4; ++i) {
        int idx = tid + i * 256;
        int r = idx >> 3, q = idx & 7;
        gaA[i] = (unsigned long long)(uintptr_t)(A  + (size_t)(bm + r) * K + q * 8);
        gaB[i] = (unsigned long long)(uintptr_t)(BT + (size_t)(bn + r) * K + q * 8);
        ldsA[i] = (unsigned)(uintptr_t)(smem + r * 64 + q * 8);           // buf0
        ldsB[i] = (unsigned)(uintptr_t)(smem + 8192 + r * 64 + q * 8);    // buf0
    }
    (void)r0; (void)q0;

    // Issue one 64-K slab (8 async ops/thread); advance K and toggle buffer.
    auto issue_stage = [&]() {
        #pragma unroll
        for (int i = 0; i < 4; ++i) async_copy_b128(ldsA[i], gaA[i]);
        #pragma unroll
        for (int i = 0; i < 4; ++i) async_copy_b128(ldsB[i], gaB[i]);
        #pragma unroll
        for (int i = 0; i < 4; ++i) {
            gaA[i] += 128; gaB[i] += 128;      // next 64-half K slab
            ldsA[i] ^= 0x8000; ldsB[i] ^= 0x8000;  // alternate 32KB buffers
        }
    };

    v8f acc[4][2] = {};

    auto compute_slab = [&](const h16* As, const h16* Bs) {
        #pragma unroll
        for (int s = 0; s < 2; ++s) {          // two 32-K sub-steps
            const int ks = s * 32;
            v16h af[4];
            #pragma unroll
            for (int mt = 0; mt < 4; ++mt) {
                int row = wm + mt * 16 + lhalf;
                int kh  = ks + lhi * 8;
                V16U u;
                u.h[0] = *(const v8h*)&As[row * 64 + kh];
                u.h[1] = *(const v8h*)&As[row * 64 + kh + 16];
                af[mt] = u.v;
            }
            v16h bf[2];
            #pragma unroll
            for (int nt = 0; nt < 2; ++nt) {
                int col = wn + nt * 16 + lhalf;
                int kb  = ks + lhi * 8;
                V16U u;
                u.h[0] = *(const v8h*)&Bs[col * 64 + kb];
                u.h[1] = *(const v8h*)&Bs[col * 64 + kb + 16];
                bf[nt] = u.v;
            }
            #pragma unroll
            for (int mt = 0; mt < 4; ++mt)
                #pragma unroll
                for (int nt = 0; nt < 2; ++nt)
                    acc[mt][nt] = __builtin_amdgcn_wmma_f32_16x16x32_f16(
                        false, af[mt], false, bf[nt],
                        (short)0, acc[mt][nt], false, false);
        }
    };

    const int nIter = K >> 6;
    issue_stage();                             // slab 0 -> buf 0

    for (int it = 0; it < nIter - 1; ++it) {   // branch-free steady state
        issue_stage();                         // slab it+1 -> other buf
        // retire slab it's 8 ops; the 8 newest stay in flight (in-order)
        asm volatile("s_wait_asynccnt 0x8" ::: "memory");
        __syncthreads();
        const h16* As = smem + (it & 1) * 16384;
        compute_slab(As, As + 8192);
        __syncthreads();                       // before buf reuse next iter
    }
    asm volatile("s_wait_asynccnt 0x0" ::: "memory");
    __syncthreads();
    {
        const h16* As = smem + ((nIter - 1) & 1) * 16384;
        compute_slab(As, As + 8192);
    }

    // ---- epilogue: VGPR r holds M=r (+8 for lanes>=16), N=lane%16 ----
    #pragma unroll
    for (int mt = 0; mt < 4; ++mt) {
        #pragma unroll
        for (int nt = 0; nt < 2; ++nt) {
            const int col = bn + wn + nt * 16 + lhalf;
            const float bcol = bias[col];
            #pragma unroll
            for (int r = 0; r < 8; ++r) {
                int row = bm + wm + mt * 16 + lhi * 8 + r;
                float v = acc[mt][nt][r] + bcol;
                if (ACT == 1) v = v * sigmoidf_(v);        // silu
                if (HAS_RES) v += res[(size_t)row * N + col];
                if (OUT_F16) ((h16*)Cout)[(size_t)row * N + col] = (h16)v;
                else         ((float*)Cout)[(size_t)row * N + col] = v;
            }
        }
    }
}

// ---------------------------------------------------------------------------
// Chunked parallel linear recurrence: state = decay*state + u_t, out = state*g_t
// ---------------------------------------------------------------------------
__global__ __launch_bounds__(256)
void scan_pass1_kernel(const float* __restrict__ U, const float* __restrict__ sd,
                       float* __restrict__ chunk_sum)
{
    const int d = blockIdx.y * 256 + threadIdx.x;
    const int c = blockIdx.x;
    const float decay = sigmoidf_(sd[d]);
    const float* up = U + (size_t)c * CHLEN * DMOD + d;
    float st = 0.f;
    #pragma unroll 4
    for (int t = 0; t < CHLEN; ++t) st = decay * st + up[(size_t)t * DMOD];
    chunk_sum[(size_t)c * DMOD + d] = st;
}

__global__ __launch_bounds__(256)
void scan_pass2_kernel(const float* __restrict__ chunk_sum,
                       const float* __restrict__ sd,
                       float* __restrict__ carry)
{
    const int d = blockIdx.x * 256 + threadIdx.x;
    const float decay = sigmoidf_(sd[d]);
    const float dp = __expf((float)CHLEN * __logf(decay));   // decay^CHLEN
    float cst = 0.f;
    for (int c = 0; c < NCHUNK; ++c) {
        carry[(size_t)c * DMOD + d] = cst;
        cst = dp * cst + chunk_sum[(size_t)c * DMOD + d];
    }
}

__global__ __launch_bounds__(256)
void scan_pass3_kernel(const float* __restrict__ U, const float* __restrict__ G,
                       const float* __restrict__ sd, const float* __restrict__ carry,
                       h16* __restrict__ S)
{
    const int d = blockIdx.y * 256 + threadIdx.x;
    const int c = blockIdx.x;
    const float decay = sigmoidf_(sd[d]);
    float st = carry[(size_t)c * DMOD + d];
    const float* up = U + (size_t)c * CHLEN * DMOD + d;
    const float* gp = G + (size_t)c * CHLEN * DMOD + d;
    h16* sp = S + (size_t)c * CHLEN * DMOD + d;
    #pragma unroll 2
    for (int t = 0; t < CHLEN; ++t) {
        st = decay * st + up[(size_t)t * DMOD];
        float gt = sigmoidf_(gp[(size_t)t * DMOD]);
        sp[(size_t)t * DMOD] = (h16)(st * gt);
    }
}

// ---------------------------------------------------------------------------
// Host-side launcher
// ---------------------------------------------------------------------------
extern "C" void kernel_launch(void* const* d_in, const int* in_sizes, int n_in,
                              void* d_out, int out_size, void* d_ws, size_t ws_size,
                              hipStream_t stream)
{
    (void)in_sizes; (void)n_in; (void)out_size; (void)ws_size;

    const size_t T = TSEQ, D = DMOD;

    const float* x      = (const float*)d_in[0];
    const float* ln1_g  = (const float*)d_in[1];
    const float* ln1_b  = (const float*)d_in[2];
    const float* W_in   = (const float*)d_in[3];
    const float* b_in   = (const float*)d_in[4];
    const float* W_gate = (const float*)d_in[5];
    const float* b_gate = (const float*)d_in[6];
    const float* W_out  = (const float*)d_in[7];
    const float* b_out  = (const float*)d_in[8];
    const float* sdecay = (const float*)d_in[9];
    const float* ln2_g  = (const float*)d_in[10];
    const float* ln2_b  = (const float*)d_in[11];
    const float* W_ff1  = (const float*)d_in[12];
    const float* b_ff1  = (const float*)d_in[13];
    const float* W_ff2  = (const float*)d_in[14];
    const float* b_ff2  = (const float*)d_in[15];
    float* out = (float*)d_out;

    // workspace bump allocator (~313 MB total)
    char* ws = (char*)d_ws;
    size_t off = 0;
    auto alloc = [&](size_t bytes) -> char* {
        char* p = ws + off;
        off += (bytes + 255) & ~(size_t)255;
        return p;
    };
    h16*   hid16   = (h16*)  alloc(T * D * 2);        // LN output / later h2
    h16*   s16     = (h16*)  alloc(T * D * 2);        // scan output (f16 A)
    float* u_f32   = (float*)alloc(T * D * 4);        // pre-scan; reused as ff16
    float* g_f32   = (float*)alloc(T * D * 4);        // gate logits
    float* x2_f32  = (float*)alloc(T * D * 4);        // post-W_out residual
    h16*   WinT    = (h16*)  alloc(D * D * 2);        // [N,K] transposed f16
    h16*   WgT     = (h16*)  alloc(D * D * 2);
    h16*   WoT     = (h16*)  alloc(D * D * 2);
    h16*   Wff1T   = (h16*)  alloc(D * 2 * D * 2);    // [2D, D]
    h16*   Wff2T   = (h16*)  alloc(2 * D * D * 2);    // [D, 2D]
    float* csum    = (float*)alloc(NCHUNK * D * 4);
    float* carry   = (float*)alloc(NCHUNK * D * 4);
    h16*   ff16    = (h16*)u_f32;                     // reuse after scan

    dim3 blk(256);

    // 1) LN1 -> f16
    ln_to_f16_kernel<<<dim3((unsigned)T), blk, 0, stream>>>(x, ln1_g, ln1_b, hid16, (int)D);

    // 2) weight convert+transpose (then resident in L2 as f16)
    transpose_to_f16_kernel<<<dim3((unsigned)(D / 32), (unsigned)(D / 32)), blk, 0, stream>>>(
        W_in, WinT, (int)D, (int)D);
    transpose_to_f16_kernel<<<dim3((unsigned)(D / 32), (unsigned)(D / 32)), blk, 0, stream>>>(
        W_gate, WgT, (int)D, (int)D);
    transpose_to_f16_kernel<<<dim3((unsigned)(D / 32), (unsigned)(D / 32)), blk, 0, stream>>>(
        W_out, WoT, (int)D, (int)D);
    transpose_to_f16_kernel<<<dim3((unsigned)(2 * D / 32), (unsigned)(D / 32)), blk, 0, stream>>>(
        W_ff1, Wff1T, (int)D, (int)(2 * D));
    transpose_to_f16_kernel<<<dim3((unsigned)(D / 32), (unsigned)(2 * D / 32)), blk, 0, stream>>>(
        W_ff2, Wff2T, (int)(2 * D), (int)D);

    // 3) u = hidden @ W_in + b_in
    gemm_wmma_kernel<0, false, false><<<dim3((unsigned)(D / 128), (unsigned)(T / 128)), blk, 0, stream>>>(
        hid16, WinT, b_in, nullptr, u_f32, (int)T, (int)D, (int)D);
    // 4) gate logits (sigmoid fused into scan)
    gemm_wmma_kernel<0, false, false><<<dim3((unsigned)(D / 128), (unsigned)(T / 128)), blk, 0, stream>>>(
        hid16, WgT, b_gate, nullptr, g_f32, (int)T, (int)D, (int)D);

    // 5) chunked diagonal scan -> s16
    scan_pass1_kernel<<<dim3(NCHUNK, (unsigned)(D / 256)), blk, 0, stream>>>(u_f32, sdecay, csum);
    scan_pass2_kernel<<<dim3((unsigned)(D / 256)), blk, 0, stream>>>(csum, sdecay, carry);
    scan_pass3_kernel<<<dim3(NCHUNK, (unsigned)(D / 256)), blk, 0, stream>>>(u_f32, g_f32, sdecay, carry, s16);

    // 6) x2 = x + s @ W_out + b_out
    gemm_wmma_kernel<0, false, true><<<dim3((unsigned)(D / 128), (unsigned)(T / 128)), blk, 0, stream>>>(
        s16, WoT, b_out, x, x2_f32, (int)T, (int)D, (int)D);

    // 7) LN2 -> f16 (reuse hid16)
    ln_to_f16_kernel<<<dim3((unsigned)T), blk, 0, stream>>>(x2_f32, ln2_g, ln2_b, hid16, (int)D);

    // 8) ff_a = silu(h2 @ W_ff1 + b_ff1) -> f16
    gemm_wmma_kernel<1, true, false><<<dim3((unsigned)(2 * D / 128), (unsigned)(T / 128)), blk, 0, stream>>>(
        hid16, Wff1T, b_ff1, nullptr, ff16, (int)T, (int)(2 * D), (int)D);

    // 9) out = x2 + ff_a @ W_ff2 + b_ff2
    gemm_wmma_kernel<0, false, true><<<dim3((unsigned)(D / 128), (unsigned)(T / 128)), blk, 0, stream>>>(
        ff16, Wff2T, b_ff2, x2_f32, out, (int)T, (int)D, (int)(2 * D));
}
